// LongSequenceAttention_49108656062986
// MI455X (gfx1250) — matrix operand order repaired and checked by reading, not compiled, not run on hardware
//
#include <hip/hip_runtime.h>
#include <hip/hip_bf16.h>

typedef __attribute__((ext_vector_type(16))) __bf16 v16bf;
typedef __attribute__((ext_vector_type(8)))  float  v8f;

#define EMBED 1024
#define NHEAD 16
#define HDIM  64
#define WIN   512
#define SEQ   8192
#define NWIN  16
#define NTOK  16384   // 2 * 8192 tokens

union Frag {
  v16bf v;
  uint4 q[2];
  unsigned u[8];
};

__device__ __forceinline__ unsigned short f2bf(float f) {
  unsigned u = __float_as_uint(f);
  u += 0x7FFFu + ((u >> 16) & 1u);   // round-to-nearest-even
  return (unsigned short)(u >> 16);
}

__device__ __forceinline__ v8f wmma_bf16(const Frag& a, const Frag& b, v8f c) {
  return __builtin_amdgcn_wmma_f32_16x16x32_bf16(false, a.v, false, b.v,
                                                 (short)0, c, false, false);
}

// ---------------- fp32 -> bf16 conversion (4 elems / thread) ----------------
__global__ void cvt_bf16_kernel(const float* __restrict__ in,
                                unsigned short* __restrict__ out, int n) {
  int i = (blockIdx.x * blockDim.x + threadIdx.x) * 4;
  if (i + 3 < n) {
    float4 f = *(const float4*)(in + i);
    uint2 o;
    o.x = (unsigned)f2bf(f.x) | ((unsigned)f2bf(f.y) << 16);
    o.y = (unsigned)f2bf(f.z) | ((unsigned)f2bf(f.w) << 16);
    *(uint2*)(out + i) = o;
  }
}

// ---------------- wave-level 32x64 bf16 GEMM tile:  Y = A * W^T -------------
// A: [M,K] bf16 row-major.  W: [N,K] bf16 row-major (torch Linear weight).
__device__ __forceinline__ void gemm_tile_32x64(
    const unsigned short* __restrict__ A, const unsigned short* __restrict__ W,
    int K, int m0, int n0, v8f acc[2][4]) {
  const int lane = threadIdx.x & 31;
  const int half = lane >> 4;
  const int r    = lane & 15;
#pragma unroll
  for (int mt = 0; mt < 2; ++mt)
#pragma unroll
    for (int nt = 0; nt < 4; ++nt)
#pragma unroll
      for (int j = 0; j < 8; ++j) acc[mt][nt][j] = 0.0f;

  const unsigned short* a0 = A + (size_t)(m0 + r) * K + 8 * half;
  const unsigned short* a1 = A + (size_t)(m0 + 16 + r) * K + 8 * half;
  const unsigned short* w0 = W + (size_t)(n0 + r) * K + 8 * half;

  for (int k0 = 0; k0 < K; k0 += 32) {
    Frag a[2], b[4];
    // A fragments (16x32 tiles): lane half 0 -> K 0..7 & 16..23, half 1 -> 8..15 & 24..31
    a[0].q[0] = *(const uint4*)(a0 + k0);
    a[0].q[1] = *(const uint4*)(a0 + k0 + 16);
    a[1].q[0] = *(const uint4*)(a1 + k0);
    a[1].q[1] = *(const uint4*)(a1 + k0 + 16);
    // B fragments (32x16 tiles of W^T): lane = out-feature column, same K split as A
#pragma unroll
    for (int nt = 0; nt < 4; ++nt) {
      const unsigned short* wp = w0 + (size_t)(nt * 16) * K + k0;
      b[nt].q[0] = *(const uint4*)(wp);
      b[nt].q[1] = *(const uint4*)(wp + 16);
    }
    __builtin_prefetch(a0 + k0 + 32, 0, 0);
    __builtin_prefetch(w0 + k0 + 32, 0, 0);
#pragma unroll
    for (int mt = 0; mt < 2; ++mt)
#pragma unroll
      for (int nt = 0; nt < 4; ++nt)
        acc[mt][nt] = wmma_bf16(a[mt], b[nt], acc[mt][nt]);
  }
}

// ---------------- QKV projection: scatter into attention-friendly layouts ---
// Q,K: per (b,win,head) row-major [512 x 64]; V: transposed [64 x 512].
__global__ void qkv_gemm_kernel(const unsigned short* __restrict__ xb,
                                const unsigned short* __restrict__ wq,
                                unsigned short* __restrict__ qb,
                                unsigned short* __restrict__ kb,
                                unsigned short* __restrict__ vtb) {
  const int gw = blockIdx.x * (blockDim.x >> 5) + (threadIdx.x >> 5);
  const int numNT = (3 * EMBED) / 64;            // 48
  const int m0 = (gw / numNT) * 32;
  const int n0 = (gw % numNT) * 64;

  v8f acc[2][4];
  gemm_tile_32x64(xb, wq, EMBED, m0, n0, acc);

  const int lane = threadIdx.x & 31, half = lane >> 4, r = lane & 15;
  const int which = n0 / EMBED;                  // 0=Q 1=K 2=V (wave-uniform)
  const int h     = (n0 % EMBED) / HDIM;         // head (wave-uniform)
#pragma unroll
  for (int mt = 0; mt < 2; ++mt) {
#pragma unroll
    for (int i = 0; i < 8; ++i) {
      const int row = m0 + mt * 16 + half * 8 + i;   // token index
      const int bb  = row >> 13;
      const int t2  = row & 8191;
      const int nw  = t2 >> 9;
      const int wp  = t2 & 511;                      // position in window
      const size_t grp = ((size_t)(bb * NWIN + nw) * NHEAD + h) * (WIN * HDIM);
#pragma unroll
      for (int nt = 0; nt < 4; ++nt) {
        const int d = nt * 16 + r;
        const float val = acc[mt][nt][i];
        if (which == 0)       qb [grp + (size_t)wp * HDIM + d] = f2bf(val * 0.125f);
        else if (which == 1)  kb [grp + (size_t)wp * HDIM + d] = f2bf(val);
        else                  vtb[grp + (size_t)d * WIN + wp]  = f2bf(val);
      }
    }
  }
}

// ---------------- windowed flash attention: 1 wave = 16 query rows ----------
__global__ void attn_kernel(const unsigned short* __restrict__ qb,
                            const unsigned short* __restrict__ kb,
                            const unsigned short* __restrict__ vtb,
                            unsigned short* __restrict__ ob) {
  __shared__ __align__(16) unsigned short Pl[8 * 16 * 32];   // per-wave P tiles
  const int wid  = threadIdx.x >> 5;
  const int gw   = blockIdx.x * 8 + wid;
  const int grp  = gw >> 5;                 // (b, win, head)
  const int q0   = (gw & 31) * 16;
  const int lane = threadIdx.x & 31, half = lane >> 4, r = lane & 15;

  const unsigned short* Q  = qb  + (size_t)grp * (WIN * HDIM);
  const unsigned short* Kp = kb  + (size_t)grp * (WIN * HDIM);
  const unsigned short* Vt = vtb + (size_t)grp * (WIN * HDIM);
  unsigned short* pl = Pl + wid * (16 * 32);

  // Q A-fragments, K-dim = 64 -> two 16x32 fragments (Q pre-scaled by 1/sqrt(d))
  Frag aq[2];
  const unsigned short* qp = Q + (size_t)(q0 + r) * HDIM + 8 * half;
  aq[0].q[0] = *(const uint4*)(qp);
  aq[0].q[1] = *(const uint4*)(qp + 16);
  aq[1].q[0] = *(const uint4*)(qp + 32);
  aq[1].q[1] = *(const uint4*)(qp + 48);

  v8f o[4];
  float m[8], l[8];
#pragma unroll
  for (int nt = 0; nt < 4; ++nt)
#pragma unroll
    for (int j = 0; j < 8; ++j) o[nt][j] = 0.0f;
#pragma unroll
  for (int i = 0; i < 8; ++i) { m[i] = -1e30f; l[i] = 0.0f; }

  for (int kc = 0; kc < WIN; kc += 32) {
    // ---- S = Q K^T for 32 keys (two 16-key tiles, two 32-wide d steps) ----
    Frag bk[2][2];
#pragma unroll
    for (int kt = 0; kt < 2; ++kt) {
      const unsigned short* kp = Kp + (size_t)(kc + kt * 16 + r) * HDIM + 8 * half;
      bk[kt][0].q[0] = *(const uint4*)(kp);
      bk[kt][0].q[1] = *(const uint4*)(kp + 16);
      bk[kt][1].q[0] = *(const uint4*)(kp + 32);
      bk[kt][1].q[1] = *(const uint4*)(kp + 48);
    }
    v8f s[2];
#pragma unroll
    for (int kt = 0; kt < 2; ++kt) {
#pragma unroll
      for (int j = 0; j < 8; ++j) s[kt][j] = 0.0f;
      s[kt] = wmma_bf16(aq[0], bk[kt][0], s[kt]);
      s[kt] = wmma_bf16(aq[1], bk[kt][1], s[kt]);
    }

    // ---- online softmax (row reductions across 16-lane halves) ----
    float mx[8];
#pragma unroll
    for (int i = 0; i < 8; ++i) mx[i] = fmaxf(s[0][i], s[1][i]);
#pragma unroll
    for (int d = 1; d < 16; d <<= 1)
#pragma unroll
      for (int i = 0; i < 8; ++i) mx[i] = fmaxf(mx[i], __shfl_xor(mx[i], d, 32));

    float sc[8], rs[8];
#pragma unroll
    for (int i = 0; i < 8; ++i) {
      const float mn = fmaxf(m[i], mx[i]);
      sc[i] = __expf(m[i] - mn);
      m[i]  = mn;
      const float p0 = __expf(s[0][i] - mn);
      const float p1 = __expf(s[1][i] - mn);
      s[0][i] = p0; s[1][i] = p1;
      rs[i] = p0 + p1;
    }
#pragma unroll
    for (int d = 1; d < 16; d <<= 1)
#pragma unroll
      for (int i = 0; i < 8; ++i) rs[i] += __shfl_xor(rs[i], d, 32);
#pragma unroll
    for (int i = 0; i < 8; ++i) l[i] = l[i] * sc[i] + rs[i];
#pragma unroll
    for (int nt = 0; nt < 4; ++nt)
#pragma unroll
      for (int i = 0; i < 8; ++i) o[nt][i] *= sc[i];

    // ---- re-stripe P (C-layout -> A-layout) through per-wave LDS tile ----
#pragma unroll
    for (int i = 0; i < 8; ++i) {
      const int row = half * 8 + i;
      pl[row * 32 + r]      = f2bf(s[0][i]);
      pl[row * 32 + 16 + r] = f2bf(s[1][i]);
    }
    asm volatile("s_wait_dscnt 0" ::: "memory");   // same-wave DS RAW
    Frag pf;
    const unsigned short* pr = pl + r * 32 + 8 * half;
    pf.q[0] = *(const uint4*)(pr);
    pf.q[1] = *(const uint4*)(pr + 16);

    // ---- O += P * V  (V pre-transposed: [64 x 512]) ----
#pragma unroll
    for (int nt = 0; nt < 4; ++nt) {
      Frag bv;
      const unsigned short* vp = Vt + (size_t)(nt * 16 + r) * WIN + kc + 8 * half;
      bv.q[0] = *(const uint4*)(vp);
      bv.q[1] = *(const uint4*)(vp + 16);
      o[nt] = wmma_bf16(pf, bv, o[nt]);
    }
  }

  // ---- normalize and scatter to [B,T,C] bf16 ----
  const int bb = grp >> 8, nw = (grp >> 4) & 15, h = grp & 15;
  float inv[8];
#pragma unroll
  for (int i = 0; i < 8; ++i) inv[i] = 1.0f / l[i];
#pragma unroll
  for (int i = 0; i < 8; ++i) {
    const int row = q0 + half * 8 + i;
    const size_t tok = (size_t)bb * SEQ + nw * WIN + row;
#pragma unroll
    for (int nt = 0; nt < 4; ++nt)
      ob[tok * EMBED + h * HDIM + nt * 16 + r] = f2bf(o[nt][i] * inv[i]);
  }
}

// ---------------- output projection GEMM, fp32 store ------------------------
__global__ void out_gemm_kernel(const unsigned short* __restrict__ ab,
                                const unsigned short* __restrict__ wo,
                                float* __restrict__ out) {
  const int gw = blockIdx.x * (blockDim.x >> 5) + (threadIdx.x >> 5);
  const int numNT = EMBED / 64;                  // 16
  const int m0 = (gw / numNT) * 32;
  const int n0 = (gw % numNT) * 64;

  v8f acc[2][4];
  gemm_tile_32x64(ab, wo, EMBED, m0, n0, acc);

  const int lane = threadIdx.x & 31, half = lane >> 4, r = lane & 15;
#pragma unroll
  for (int mt = 0; mt < 2; ++mt)
#pragma unroll
    for (int i = 0; i < 8; ++i) {
      const int row = m0 + mt * 16 + half * 8 + i;
#pragma unroll
      for (int nt = 0; nt < 4; ++nt)
        out[(size_t)row * EMBED + n0 + nt * 16 + r] = acc[mt][nt][i];
    }
}

extern "C" void kernel_launch(void* const* d_in, const int* in_sizes, int n_in,
                              void* d_out, int out_size, void* d_ws, size_t ws_size,
                              hipStream_t stream) {
  (void)in_sizes; (void)n_in; (void)out_size; (void)ws_size;
  const float* x     = (const float*)d_in[0];
  const float* w_qkv = (const float*)d_in[1];
  const float* w_out = (const float*)d_in[2];
  float* out = (float*)d_out;

  char* ws = (char*)d_ws;
  // bf16 staging buffers (byte offsets); attn output aliases xb (x consumed by then)
  unsigned short* xb    = (unsigned short*)(ws);               //  33,554,432 B
  unsigned short* wqkvb = (unsigned short*)(ws +  33554432);   //   6,291,456 B
  unsigned short* woutb = (unsigned short*)(ws +  39845888);   //   2,097,152 B
  unsigned short* qb    = (unsigned short*)(ws +  41943040);   //  33,554,432 B
  unsigned short* kb    = (unsigned short*)(ws +  75497472);   //  33,554,432 B
  unsigned short* vtb   = (unsigned short*)(ws + 109051904);   //  33,554,432 B
  unsigned short* atb   = xb;                                  // alias

  cvt_bf16_kernel<<<16384, 256, 0, stream>>>(x,     xb,    NTOK * EMBED);
  cvt_bf16_kernel<<< 3072, 256, 0, stream>>>(w_qkv, wqkvb, 3 * EMBED * EMBED);
  cvt_bf16_kernel<<< 1024, 256, 0, stream>>>(w_out, woutb, EMBED * EMBED);

  // (16384/32) * (3072/64) = 24576 waves -> 3072 blocks of 8 waves
  qkv_gemm_kernel<<<3072, 256, 0, stream>>>(xb, wqkvb, qb, kb, vtb);

  // 512 (b,win,head) groups * 32 query tiles = 16384 waves -> 2048 blocks
  attn_kernel<<<2048, 256, 0, stream>>>(qb, kb, vtb, atb);

  // (16384/32) * (1024/64) = 8192 waves -> 1024 blocks
  out_gemm_kernel<<<1024, 256, 0, stream>>>(atb, woutb, out);
}
